// SpokeLayer_19155554140668
// MI455X (gfx1250) — compile-verified, matrix operand order checked by reading, not run on hardware
//
#include <hip/hip_runtime.h>
#include <hip/hip_bf16.h>

// ---------------- problem constants ----------------
#define D_MODEL    2048
#define NUM_SPOKES 8
#define RANK       64
#define N1         (NUM_SPOKES * RANK)   // 512 total rank
#define TOKENS     (4 * 4096)            // 16384
#define TM         64                    // tokens per workgroup
#define KC         256                   // A K-tile staged in LDS
#define NKT        (D_MODEL / KC)        // 8 K-tiles
#define LDSA_STRIDE (KC + 8)             // 264 bf16 -> 16B aligned rows, 4-bank skew
#define LDSV_STRIDE (N1 + 8)             // 520 bf16
#define EPS_F      1.1920928955078125e-7f

// LDS byte offsets
#define OFF_INV  0                         // 64 floats (256 B)
#define RAWBYTES (TM * KC * 4)             // 65536 B per raw buffer
#define OFF_RAW0 256
#define OFF_RAW1 (OFF_RAW0 + RAWBYTES)     // 65792
#define OFF_A    (OFF_RAW1 + RAWBYTES)     // 131328
#define OFF_V    (OFF_A + TM * LDSA_STRIDE * 2)  // 165120
#define SMEM_BYTES (OFF_V + TM * LDSV_STRIDE * 2) // 231680  (< 320 KB/WGP)

typedef __attribute__((ext_vector_type(16))) __bf16 v16bf;
typedef __attribute__((ext_vector_type(8)))  __bf16 v8bf;
typedef __attribute__((ext_vector_type(4)))  __bf16 v4bf;
typedef __attribute__((ext_vector_type(8)))  float  v8f;
typedef unsigned int v4u __attribute__((ext_vector_type(4)));
typedef int          v8i __attribute__((ext_vector_type(8)));
typedef int          v4i __attribute__((ext_vector_type(4)));

#if __has_builtin(__builtin_amdgcn_tensor_load_to_lds) && __has_builtin(__builtin_amdgcn_s_wait_tensorcnt)
#define USE_TDM 1
#else
#define USE_TDM 0
#endif

static __device__ __forceinline__ v16bf cat8(v8bf lo, v8bf hi) {
  return __builtin_shufflevector(lo, hi, 0,1,2,3,4,5,6,7,8,9,10,11,12,13,14,15);
}

static __device__ __forceinline__ v8f wmma_bf16(v16bf a, v16bf b, v8f c) {
  return __builtin_amdgcn_wmma_f32_16x16x32_bf16(false, a, false, b, (short)0, c,
                                                 false, false);
}

#if USE_TDM
// Issue one TDM 2D tile load: [TM rows x KC fp32 cols], row stride D_MODEL elems.
// D# bitfields per CDNA5 ISA ch.8 (group0: flags/lds/global/type, group1: dims).
// Toolchain declares the 6-arg builtin: (v4u g0, v8i g1, v4i g2, v4i g3, v8i, i32 cpol).
static __device__ __forceinline__ void tdm_load_tile(const float* gtile, unsigned lds_off) {
  const unsigned long long ga = (unsigned long long)(uintptr_t)gtile;
  v4u g0;
  g0[0] = 1u;                                            // count=1, user descriptor
  g0[1] = lds_off;                                       // lds_addr (bytes)
  g0[2] = (unsigned)ga;                                  // global_addr[31:0]
  g0[3] = (unsigned)((ga >> 32) & 0x1FFFFFFu) | 0x80000000u; // ga[56:32] | type=2
  v8i g1;
  g1[0] = (int)(2u << 16);                               // workgroup_mask=0, data_size=4B
  g1[1] = (int)((unsigned)(D_MODEL & 0xFFFF) << 16);     // tensor_dim0 lo16
  g1[2] = (int)((D_MODEL >> 16) | ((TOKENS & 0xFFFF) << 16)); // td0 hi | td1 lo
  g1[3] = (int)((TOKENS >> 16) | (KC << 16));            // td1 hi | tile_dim0=KC
  g1[4] = (int)TM;                                       // tile_dim1=TM, tile_dim2=0
  g1[5] = (int)D_MODEL;                                  // tensor_dim0_stride lo32
  g1[6] = 0;
  g1[7] = 0;
  const v4i gz4 = {0, 0, 0, 0};                          // groups 2/3: dims 3/4 unused
  const v8i gz8 = {0, 0, 0, 0, 0, 0, 0, 0};
  __builtin_amdgcn_tensor_load_to_lds(g0, g1, gz4, gz4, gz8, 0);
}
#endif

// ---------------- weight prep: fp32 -> bf16, K-contiguous columns ----------------
__global__ void k_conv_w1(const float* __restrict__ src, __bf16* __restrict__ dst, int n) {
  int i = blockIdx.x * blockDim.x + threadIdx.x;
  if (i < n) dst[i] = (__bf16)src[i];
}

__global__ void k_conv_w2(const float* __restrict__ wup, __bf16* __restrict__ dst) {
  int o = blockIdx.x * blockDim.x + threadIdx.x;
  if (o >= D_MODEL * N1) return;
  int d  = o >> 9;
  int rg = o & 511;
  int s  = rg >> 6;
  int r  = rg & 63;
  dst[o] = (__bf16)wup[(size_t)s * (D_MODEL * RANK) + (size_t)d * RANK + r];
}

// ---------------- fused RMSNorm -> GEMM1 -> SiLU -> GEMM2 -> residual ----------------
__global__ __launch_bounds__(256, 1)
void k_spoke_fused(const float* __restrict__ h,
                   const __bf16* __restrict__ W1,   // [512][2048] K-contig cols
                   const __bf16* __restrict__ W2,   // [2048][512] K-contig cols
                   const float* __restrict__ gate_bias,
                   float* __restrict__ out) {
  const int tid       = threadIdx.x;
  const int lane      = tid & 31;
  const int wave      = tid >> 5;          // 0..7, owns N-slice of 64 (G1) / 256 (G2)
  const int laneLo    = lane & 15;
  const int laneHiOff = (lane >> 4) << 3;  // 0 or 8
  const int blockTok0 = blockIdx.x * TM;

  extern __shared__ char smem[];
  float*  lds_inv = (float*)(smem + OFF_INV);
  __bf16* lds_a   = (__bf16*)(smem + OFF_A);
  __bf16* lds_v   = (__bf16*)(smem + OFF_V);

  const float gb    = gate_bias[0];
  const float scale = (1.0f / (1.0f + __expf(-gb))) * (1.0f / NUM_SPOKES);

#if USE_TDM
  if (wave == 0) {  // kick off first A tile while everyone computes sums of squares
    tdm_load_tile(h + (size_t)blockTok0 * D_MODEL, OFF_RAW0);
  }
#endif

  // ---- Phase A: per-token inv_rms (4 lanes per token) ----
  {
    const int token = tid >> 2;            // 0..63
    const int part  = tid & 3;
    const float* hrow = h + (size_t)(blockTok0 + token) * D_MODEL;
    float ss = 0.0f;
    #pragma unroll 8
    for (int j = 0; j < D_MODEL / 16; ++j) {
      const float4 hv = *(const float4*)(hrow + part * 4 + j * 16);
      ss += hv.x * hv.x + hv.y * hv.y + hv.z * hv.z + hv.w * hv.w;
    }
    ss += __shfl_xor(ss, 1, 32);
    ss += __shfl_xor(ss, 2, 32);
    if (part == 0) lds_inv[token] = rsqrtf(ss * (1.0f / D_MODEL) + EPS_F);
  }
  __syncthreads();

  // ---- Phase B: GEMM1 (view = A_norm @ W1), K staged in LDS tiles of KC ----
  v8f acc[4][4] = {};
  for (int kt = 0; kt < NKT; ++kt) {
    const float* raw = (const float*)(smem + OFF_RAW0 + (kt & 1) * RAWBYTES);

#if USE_TDM
    if (wave == 0) {
      if (kt + 1 < NKT) {   // prefetch next tile into the other buffer, then wait tile kt
        tdm_load_tile(h + (size_t)blockTok0 * D_MODEL + (kt + 1) * KC,
                      (unsigned)(OFF_RAW0 + ((kt + 1) & 1) * RAWBYTES));
        __builtin_amdgcn_s_wait_tensorcnt(1);
      } else {
        __builtin_amdgcn_s_wait_tensorcnt(0);
      }
    }
    __syncthreads();
#else
    {   // cooperative fallback copy of tile kt (through VGPRs)
      float* rawW = (float*)(smem + OFF_RAW0 + (kt & 1) * RAWBYTES);
      const float* gsrc = h + (size_t)blockTok0 * D_MODEL + kt * KC;
      #pragma unroll
      for (int i = 0; i < (TM * KC / 4) / 256; ++i) {   // 16 float4 per thread
        const int seg = tid + i * 256;
        const int r   = seg >> 6;          // KC/4 = 64 segments per row
        const int cc  = (seg & 63) * 4;
        *(float4*)&rawW[r * KC + cc] = *(const float4*)(gsrc + (size_t)r * D_MODEL + cc);
      }
      __syncthreads();
    }
#endif

    // normalize + bf16 convert: each wave owns 8 rows; lanes sweep a row (no conflicts)
    #pragma unroll
    for (int j = 0; j < 8; ++j) {
      const int row    = wave * 8 + j;
      const float invr = lds_inv[row];
      #pragma unroll
      for (int c = 0; c < 2; ++c) {
        const int col = lane * 4 + c * 128;
        const float4 rv = *(const float4*)&raw[row * KC + col];
        v4bf pv;
        pv[0] = (__bf16)(rv.x * invr);
        pv[1] = (__bf16)(rv.y * invr);
        pv[2] = (__bf16)(rv.z * invr);
        pv[3] = (__bf16)(rv.w * invr);
        *(v4bf*)&lds_a[row * LDSA_STRIDE + col] = pv;
      }
    }
    __syncthreads();

    // 4 M-tiles x 4 N-tiles per wave -> 16 WMMA per 32-K step, B loaded once per wave
    for (int k0 = 0; k0 < KC; k0 += 32) {
      const int kb = k0 + laneHiOff;
      v16bf af[4];
      #pragma unroll
      for (int mi = 0; mi < 4; ++mi) {
        const __bf16* ap = &lds_a[(mi * 16 + laneLo) * LDSA_STRIDE + kb];
        af[mi] = cat8(*(const v8bf*)ap, *(const v8bf*)(ap + 16));
      }
      #pragma unroll
      for (int ni = 0; ni < 4; ++ni) {
        const int n = wave * 64 + ni * 16 + laneLo;
        const __bf16* bp = W1 + (size_t)n * D_MODEL + kt * KC + kb;
        __builtin_prefetch(bp + 32, 0, 1);
        const v16bf bfr = cat8(*(const v8bf*)bp, *(const v8bf*)(bp + 16));
        #pragma unroll
        for (int mi = 0; mi < 4; ++mi)
          acc[mi][ni] = wmma_bf16(af[mi], bfr, acc[mi][ni]);
      }
    }
    __syncthreads();   // lds_a + raw buffer free for the next tile
  }

  // ---- SiLU + bf16 store of view tile [64 x 512] to LDS ----
  #pragma unroll
  for (int ni = 0; ni < 4; ++ni) {
    const int col = wave * 64 + ni * 16 + laneLo;
    #pragma unroll
    for (int mi = 0; mi < 4; ++mi) {
      #pragma unroll
      for (int v = 0; v < 8; ++v) {
        const float x = acc[mi][ni][v];
        const float s = x / (1.0f + __expf(-x));
        lds_v[(mi * 16 + v + laneHiOff) * LDSV_STRIDE + col] = (__bf16)s;
      }
    }
  }
  __syncthreads();

  // ---- Phase C: GEMM2 (update = view @ W2), wave owns 256 d-cols in 4 chunks ----
  for (int c = 0; c < 4; ++c) {
    v8f acc2[4][4] = {};
    const int nbase = wave * 256 + c * 64;
    for (int k0 = 0; k0 < N1; k0 += 32) {
      const int kb = k0 + laneHiOff;
      v16bf af[4];
      #pragma unroll
      for (int mi = 0; mi < 4; ++mi) {
        const __bf16* ap = &lds_v[(mi * 16 + laneLo) * LDSV_STRIDE + kb];
        af[mi] = cat8(*(const v8bf*)ap, *(const v8bf*)(ap + 16));
      }
      #pragma unroll
      for (int ni = 0; ni < 4; ++ni) {
        const int dcol = nbase + ni * 16 + laneLo;
        const __bf16* bp = W2 + (size_t)dcol * N1 + kb;
        __builtin_prefetch(bp + 32, 0, 1);
        const v16bf bfr = cat8(*(const v8bf*)bp, *(const v8bf*)(bp + 16));
        #pragma unroll
        for (int mi = 0; mi < 4; ++mi)
          acc2[mi][ni] = wmma_bf16(af[mi], bfr, acc2[mi][ni]);
      }
    }
    // epilogue: out = h + sigmoid(gate)/8 * update   (coalesced b32 across laneLo)
    #pragma unroll
    for (int ni = 0; ni < 4; ++ni) {
      const int dcol = nbase + ni * 16 + laneLo;
      #pragma unroll
      for (int mi = 0; mi < 4; ++mi) {
        #pragma unroll
        for (int v = 0; v < 8; ++v) {
          const int row = blockTok0 + mi * 16 + v + laneHiOff;
          const size_t idx = (size_t)row * D_MODEL + dcol;
          out[idx] = h[idx] + scale * acc2[mi][ni][v];
        }
      }
    }
  }
}

// ---------------- host launcher ----------------
extern "C" void kernel_launch(void* const* d_in, const int* in_sizes, int n_in,
                              void* d_out, int out_size, void* d_ws, size_t ws_size,
                              hipStream_t stream) {
  (void)in_sizes; (void)n_in; (void)out_size; (void)ws_size;
  const float* h  = (const float*)d_in[0];   // [4,4096,2048] f32
  const float* wd = (const float*)d_in[1];   // [8,64,2048]   f32
  const float* wu = (const float*)d_in[2];   // [8,2048,64]   f32
  const float* gb = (const float*)d_in[3];   // scalar        f32
  float* out = (float*)d_out;

  __bf16* W1 = (__bf16*)d_ws;                       // 2 MB
  __bf16* W2 = W1 + (size_t)N1 * D_MODEL;           // 2 MB

  const int nw = N1 * D_MODEL;
  k_conv_w1<<<(nw + 255) / 256, 256, 0, stream>>>(wd, W1, nw);
  k_conv_w2<<<(nw + 255) / 256, 256, 0, stream>>>(wu, W2);

  static bool attr_set = []() {
    (void)hipFuncSetAttribute((const void*)k_spoke_fused,
                              hipFuncAttributeMaxDynamicSharedMemorySize, SMEM_BYTES);
    return true;
  }();
  (void)attr_set;

  k_spoke_fused<<<TOKENS / TM, 256, (size_t)SMEM_BYTES, stream>>>(h, W1, W2, gb, out);
}
